// ElectronicStructureConvLayer_71519795413189
// MI455X (gfx1250) — matrix operand order repaired and checked by reading, not compiled, not run on hardware
//
#include <hip/hip_runtime.h>
#include <hip/hip_bf16.h>
#include <stdint.h>

typedef __bf16 bf16;
typedef bf16  v16bf __attribute__((ext_vector_type(16)));
typedef bf16  v8bf  __attribute__((ext_vector_type(8)));
typedef float v8f   __attribute__((ext_vector_type(8)));
typedef unsigned int v4u __attribute__((ext_vector_type(4)));
typedef int  v8i __attribute__((ext_vector_type(8)));
typedef int  v4i __attribute__((ext_vector_type(4)));

#define NNODES 4096
#define HDIM   256
#define NEDGES 131072
#define NHEADS 8
#define DH     32

#if defined(__gfx1250__) && __has_builtin(__builtin_amdgcn_tensor_load_to_lds)
#define USE_TDM 1
#else
#define USE_TDM 0
#endif

#if defined(__gfx1250__) && __has_builtin(__builtin_amdgcn_exp2f)
#define FAST_EXP2(x) __builtin_amdgcn_exp2f(x)
#else
#define FAST_EXP2(x) exp2f(x)
#endif

__device__ __forceinline__ v8bf ld8(const bf16* p) {
  return *reinterpret_cast<const v8bf*>(p);
}
__device__ __forceinline__ v16bf mk16(v8bf lo, v8bf hi) {
  v16bf r;
#pragma unroll
  for (int i = 0; i < 8; ++i) { r[i] = lo[i]; r[i + 8] = hi[i]; }
  return r;
}
__device__ __forceinline__ v8f wmma_bf16(v16bf a, v16bf b, v8f c) {
  // D = A(16x32) * B(32x16) + C, fp32 accumulate
  return __builtin_amdgcn_wmma_f32_16x16x32_bf16(false, a, false, b, (short)0, c,
                                                 false, false);
}

// ---------------------------------------------------------------------------
// Stage a contiguous [64 x K] bf16 slab of Wt into LDS.
// On gfx1250 this uses the Tensor Data Mover: wave 0 issues TENSOR_LOAD_TO_LDS
// with a 2D descriptor (tile_dim0 = K elements contiguous, tile_dim1 = 64 rows,
// dim0 stride = K), waits on TENSORcnt, then the workgroup barrier releases
// the other waves.  Fallback: cooperative vector copy.
// ---------------------------------------------------------------------------
__device__ __forceinline__ void stage_w_lds(bf16* Wlds, const bf16* Wslab,
                                            int K) {
#if USE_TDM
  if (threadIdx.x < 32) {
    const uint64_t ga = (uint64_t)(uintptr_t)Wslab;          // global byte addr
    const uint32_t la = (uint32_t)(uintptr_t)Wlds;           // LDS byte offset
    v4u g0;
    g0[0] = 1u;                                   // count=1 (valid), user mode
    g0[1] = la;                                   // lds_addr [63:32]
    g0[2] = (uint32_t)ga;                         // global_addr [95:64]
    g0[3] = (uint32_t)(ga >> 32) | (2u << 30);    // addr [120:96] | type=2
    v8i g1;
    g1[0] = 1 << 16;                              // data_size=1 (2 bytes/elem)
    g1[1] = (K & 0xFFFF) << 16;                   // tensor_dim0[15:0]
    g1[2] = 64 << 16;                             // tensor_dim1 = 64 rows
    g1[3] = (K & 0xFFFF) << 16;                   // tile_dim0 = K
    g1[4] = 64;                                   // tile_dim1 = 64
    g1[5] = K;                                    // tensor_dim0_stride = K
    g1[6] = 0;
    g1[7] = 0;
    const v4i gz4 = {0, 0, 0, 0};
    const v8i gz8 = {0, 0, 0, 0, 0, 0, 0, 0};
    __builtin_amdgcn_tensor_load_to_lds(g0, g1, gz4, gz4, gz8, 0);
    __builtin_amdgcn_s_wait_tensorcnt(0);
  }
  __syncthreads();
#else
  const int nvec = (64 * K) / 8;
  for (int i = threadIdx.x; i < nvec; i += 256)
    reinterpret_cast<v8bf*>(Wlds)[i] = reinterpret_cast<const v8bf*>(Wslab)[i];
  __syncthreads();
#endif
}

// ---------------------------------------------------------------------------
// Generic bf16 WMMA GEMM: Out(MxN=256) = epi(A(MxK) @ W(KxN) + bias), W given
// pre-transposed bf16 as Wt[N][K].  Weight slab for the block's 64 columns is
// staged in LDS (TDM); B fragments are ds_load_b128 pairs.
// Block: 256 thr = 8 waves; block tile 128x64; wave tile 32x32 (2x2 wmma frags)
// AMODE: 0 = plain A0 (row stride K)
//        1 = edge gather: row e -> [att[src[e]] (K 0..255) | att[dst[e]] (256..511)]
//        2 = concat: row r -> [A0[r] (0..255) | A1[r] (256..511)]
// EPI:   0 = bf16 store        1 = bf16 relu store   2 = f32 store
//        3 = atomicAdd f32 into Out[dst[row]*256+col] (segment sum)
//        4 = scatter bf16 to [h][n][d] (h=col>>5, d=col&31)  (q/k layout)
//        5 = scatter bf16 to [col][row] (vT layout, col-major)
// ---------------------------------------------------------------------------
template <int AMODE, int EPI>
__global__ __launch_bounds__(256) void gemm_k(
    const bf16* __restrict__ A0, const bf16* __restrict__ A1,
    const int* __restrict__ eidx, const bf16* __restrict__ Wt,
    const float* __restrict__ bias, void* __restrict__ Out, int M, int K,
    float escale) {
  extern __shared__ __align__(16) char smem[];
  bf16* Wlds = reinterpret_cast<bf16*>(smem);

  const int lane = threadIdx.x & 31;
  const int l16  = lane & 15;
  const int up   = lane >> 4;
  const int w    = threadIdx.x >> 5;
  const int rowBase = blockIdx.x * 128 + (w & 3) * 32;
  const int colLoc  = (w >> 2) * 32;                 // column offset inside LDS slab
  const int colBase = blockIdx.y * 64 + colLoc;
  const int off0 = up ? 8 : 0;

  stage_w_lds(Wlds, Wt + (size_t)blockIdx.y * 64 * K, K);

  int es[2], ed[2];
  if constexpr (AMODE == 1) {
#pragma unroll
    for (int i = 0; i < 2; ++i) {
      const int e = rowBase + i * 16 + l16;
      es[i] = eidx[2 * e];
      ed[i] = eidx[2 * e + 1];
    }
  }

  v8f c[2][2] = {};
  for (int kb = 0; kb < K; kb += 32) {
    v16bf a[2], b[2];
#pragma unroll
    for (int i = 0; i < 2; ++i) {
      const bf16* base;
      if constexpr (AMODE == 0) {
        base = A0 + (size_t)(rowBase + i * 16 + l16) * K + kb;
      } else if constexpr (AMODE == 1) {
        base = (kb < 256) ? (A0 + (size_t)es[i] * 256 + kb)
                          : (A0 + (size_t)ed[i] * 256 + (kb - 256));
      } else {
        const int r = rowBase + i * 16 + l16;
        base = (kb < 256) ? (A0 + (size_t)r * 256 + kb)
                          : (A1 + (size_t)r * 256 + (kb - 256));
      }
      // A layout: lane<16 -> K chunks {kb+0..7, kb+16..23}; lane>=16 -> {+8,+24}
      a[i] = mk16(ld8(base + off0), ld8(base + off0 + 16));
      if (kb + 32 < K) __builtin_prefetch(base + 32, 0, 1);
    }
    const int kc = kb + (up ? 8 : 0);
#pragma unroll
    for (int j = 0; j < 2; ++j) {
      // B layout: lane<16 -> col, K chunks {kb+0..7, kb+16..23}; lane>=16 -> {+8,+24}
      const bf16* bp = Wlds + (size_t)(colLoc + j * 16 + l16) * K + kc;
      b[j] = mk16(ld8(bp), ld8(bp + 16));
    }
#pragma unroll
    for (int i = 0; i < 2; ++i)
#pragma unroll
      for (int j = 0; j < 2; ++j) c[i][j] = wmma_bf16(a[i], b[j], c[i][j]);
  }

  // Epilogue. C layout: reg r, lane<16 -> (row rb+r, col), lane>=16 -> (rb+8+r)
#pragma unroll
  for (int j = 0; j < 2; ++j) {
    const int col = colBase + j * 16 + l16;
    const float bj = bias[col];
#pragma unroll
    for (int i = 0; i < 2; ++i) {
      const int rb = rowBase + i * 16 + up * 8;
#pragma unroll
      for (int r = 0; r < 8; ++r) {
        float v = (c[i][j][r] + bj) * escale;
        if constexpr (EPI == 1) v = fmaxf(v, 0.0f);
        const int row = rb + r;
        if constexpr (EPI == 0 || EPI == 1) {
          ((bf16*)Out)[(size_t)row * 256 + col] = (bf16)v;
        } else if constexpr (EPI == 2) {
          ((float*)Out)[(size_t)row * 256 + col] = v;
        } else if constexpr (EPI == 3) {
          const int d = eidx[2 * row + 1];
          atomicAdd(((float*)Out) + (size_t)d * 256 + col, v);
        } else if constexpr (EPI == 4) {
          ((bf16*)Out)[((size_t)(col >> 5) * NNODES + row) * DH + (col & 31)] =
              (bf16)v;
        } else {  // EPI == 5 : vT[col][row]
          ((bf16*)Out)[(size_t)col * NNODES + row] = (bf16)v;
        }
      }
    }
  }
}

// ---------------------------------------------------------------------------
// Flash attention, transposed-score formulation.
// q,k stored bf16 [h][n][dh] (q pre-scaled by log2e/sqrt(dh)), v stored
// transposed bf16 [h][dh][n]. Each wave owns 16 queries of one head, loops
// over keys in 32-key tiles:
//   S^T(32k x 16q) = K_tile(16x32dh) @ Q^T   -> 2 wmma
//   online softmax per query = per lane pair (shfl_xor 16)
//   O^T(32d x 16q) += V^T_tile @ P^T         -> 2 wmma
// Output attpre bf16 [n][h*32+d].
// ---------------------------------------------------------------------------
__global__ __launch_bounds__(256) void attn_k(const bf16* __restrict__ q,
                                              const bf16* __restrict__ k,
                                              const bf16* __restrict__ vT,
                                              bf16* __restrict__ attpre) {
  const int lane = threadIdx.x & 31;
  const int l16  = lane & 15;
  const int up   = lane >> 4;
  const int h    = blockIdx.y;
  const int qb   = blockIdx.x * 128 + (threadIdx.x >> 5) * 16;

  const bf16* qh = q + (size_t)h * NNODES * DH;
  const bf16* kh = k + (size_t)h * NNODES * DH;
  const bf16* vh = vT + (size_t)h * DH * NNODES;

  // B fragment of Q^T: lane<16 -> query qb+l16, dh 0..15; lane>=16 -> dh 16..31
  const bf16* qp = qh + (size_t)(qb + l16) * DH + (up ? 16 : 0);
  const v16bf Bq = mk16(ld8(qp), ld8(qp + 8));

  v8f Ot0 = {}, Ot1 = {};
  float m = -3.0e38f, lsum = 0.0f;
  const int off0 = up ? 16 : 0;

  for (int kb = 0; kb < NNODES; kb += 32) {
    // A fragments of K tile (rows = keys, Kdim = dh = 32)
    const bf16* kp0 = kh + (size_t)(kb + l16) * DH + off0;
    const bf16* kp1 = kh + (size_t)(kb + 16 + l16) * DH + off0;
    const v16bf Ak0 = mk16(ld8(kp0), ld8(kp0 + 8));
    const v16bf Ak1 = mk16(ld8(kp1), ld8(kp1 + 8));
    v8f z = {};
    v8f St0 = wmma_bf16(Ak0, Bq, z);  // keys kb..kb+15  x 16 queries
    v8f St1 = wmma_bf16(Ak1, Bq, z);  // keys kb+16..+31 x 16 queries

    float mloc = St0[0];
#pragma unroll
    for (int r = 0; r < 8; ++r) {
      mloc = fmaxf(mloc, St0[r]);
      mloc = fmaxf(mloc, St1[r]);
    }
    mloc = fmaxf(mloc, __shfl_xor(mloc, 16, 32));
    const float mnew  = fmaxf(m, mloc);
    const float alpha = FAST_EXP2(m - mnew);

    float p0[8], p1[8], ssum = 0.0f;
#pragma unroll
    for (int r = 0; r < 8; ++r) {
      p0[r] = FAST_EXP2(St0[r] - mnew);
      p1[r] = FAST_EXP2(St1[r] - mnew);
      ssum += p0[r] + p1[r];
    }
    ssum += __shfl_xor(ssum, 16, 32);
    lsum = lsum * alpha + ssum;
    m = mnew;
#pragma unroll
    for (int r = 0; r < 8; ++r) {
      Ot0[r] *= alpha;
      Ot1[r] *= alpha;
    }

    // P^T C-layout -> B fragment: one lane-pair swap per register
    v16bf Bp;
#pragma unroll
    for (int r = 0; r < 8; ++r) {
      const float t0 = __shfl_xor(p0[r], 16, 32);
      const float t1 = __shfl_xor(p1[r], 16, 32);
      Bp[r]     = (bf16)(up ? t1 : p0[r]);  // key kb+16*up+r of this query
      Bp[r + 8] = (bf16)(up ? p1[r] : t0);  // key kb+16*up+8+r
    }

    // A fragments of V^T tile (rows = dh, Kdim = 32 keys), contiguous in vT
    const bf16* vp0 = vh + (size_t)l16 * NNODES + kb + off0;
    const bf16* vp1 = vh + (size_t)(l16 + 16) * NNODES + kb + off0;
    const v16bf Av0 = mk16(ld8(vp0), ld8(vp0 + 8));
    const v16bf Av1 = mk16(ld8(vp1), ld8(vp1 + 8));
    Ot0 = wmma_bf16(Av0, Bp, Ot0);
    Ot1 = wmma_bf16(Av1, Bp, Ot1);
  }

  const float inv = 1.0f / lsum;
  const int qrow  = qb + l16;
  const int dbase = up * 8;
  v8bf o0, o1;
#pragma unroll
  for (int r = 0; r < 8; ++r) {
    o0[r] = (bf16)(Ot0[r] * inv);
    o1[r] = (bf16)(Ot1[r] * inv);
  }
  *reinterpret_cast<v8bf*>(attpre + (size_t)qrow * HDIM + h * DH + dbase) = o0;
  *reinterpret_cast<v8bf*>(attpre + (size_t)qrow * HDIM + h * DH + 16 + dbase) = o1;
}

// ---------------------------------------------------------------------------
// Small utility kernels
// ---------------------------------------------------------------------------
__global__ void transpose_w_k(const float* __restrict__ W, bf16* __restrict__ Wt,
                              int K, int Nn) {
  const int idx = blockIdx.x * 256 + threadIdx.x;
  if (idx >= K * Nn) return;
  const int kk = idx / Nn, nn = idx - kk * Nn;
  Wt[(size_t)nn * K + kk] = (bf16)W[idx];
}
__global__ void cvt_bf16_k(const float* __restrict__ s, bf16* __restrict__ d,
                           int n) {
  const int i = blockIdx.x * 256 + threadIdx.x;
  if (i < n) d[i] = (bf16)s[i];
}
__global__ void zero_f32_k(float* __restrict__ p, int n) {
  const int i = blockIdx.x * 256 + threadIdx.x;
  if (i < n) p[i] = 0.0f;
}

// ---------------------------------------------------------------------------
extern "C" void kernel_launch(void* const* d_in, const int* in_sizes, int n_in,
                              void* d_out, int out_size, void* d_ws,
                              size_t ws_size, hipStream_t stream) {
  (void)in_sizes; (void)n_in; (void)out_size; (void)ws_size;
  const float* x   = (const float*)d_in[0];
  const int*   ei  = (const int*)d_in[1];
  const float* Wq  = (const float*)d_in[2];
  const float* Wk  = (const float*)d_in[3];
  const float* Wv  = (const float*)d_in[4];
  const float* bq  = (const float*)d_in[5];
  const float* bk  = (const float*)d_in[6];
  const float* bv  = (const float*)d_in[7];
  const float* Wo  = (const float*)d_in[8];
  const float* bo  = (const float*)d_in[9];
  const float* Wm1 = (const float*)d_in[10];
  const float* bm1 = (const float*)d_in[11];
  const float* Wm2 = (const float*)d_in[12];
  const float* bm2 = (const float*)d_in[13];
  const float* Wu1 = (const float*)d_in[14];
  const float* bu1 = (const float*)d_in[15];
  const float* Wu2 = (const float*)d_in[16];
  const float* bu2 = (const float*)d_in[17];

  uint8_t* p = (uint8_t*)d_ws;
  auto carve = [&](size_t bytes) {
    uint8_t* r = p;
    p += (bytes + 255) & ~(size_t)255;
    return r;
  };
  bf16* xbf    = (bf16*)carve((size_t)NNODES * HDIM * 2);
  bf16* Wqt    = (bf16*)carve((size_t)HDIM * HDIM * 2);
  bf16* Wkt    = (bf16*)carve((size_t)HDIM * HDIM * 2);
  bf16* Wvt    = (bf16*)carve((size_t)HDIM * HDIM * 2);
  bf16* Wot    = (bf16*)carve((size_t)HDIM * HDIM * 2);
  bf16* Wm1t   = (bf16*)carve((size_t)2 * HDIM * HDIM * 2);
  bf16* Wm2t   = (bf16*)carve((size_t)HDIM * HDIM * 2);
  bf16* Wu1t   = (bf16*)carve((size_t)2 * HDIM * HDIM * 2);
  bf16* Wu2t   = (bf16*)carve((size_t)HDIM * HDIM * 2);
  bf16* qbuf   = (bf16*)carve((size_t)NHEADS * NNODES * DH * 2);
  bf16* kbuf   = (bf16*)carve((size_t)NHEADS * NNODES * DH * 2);
  bf16* vTbuf  = (bf16*)carve((size_t)NHEADS * DH * NNODES * 2);
  bf16* attpre = (bf16*)carve((size_t)NNODES * HDIM * 2);
  bf16* att    = (bf16*)carve((size_t)NNODES * HDIM * 2);
  float* msg   = (float*)carve((size_t)NNODES * HDIM * 4);
  bf16* msgbf  = (bf16*)carve((size_t)NNODES * HDIM * 2);
  bf16* hid    = (bf16*)carve((size_t)NEDGES * HDIM * 2);  // edge/upd hidden

  const dim3 blk(256);
  const size_t smem256 = 64 * 256 * sizeof(bf16);  // 32 KB weight slab (K=256)
  const size_t smem512 = 64 * 512 * sizeof(bf16);  // 64 KB weight slab (K=512)
  // fold log2(e)/sqrt(dh) into q so softmax uses exp2 (v_exp_f32)
  const float QSCALE = 1.4426950408889634f * 0.17677669529663687f;

  // --- precision prep -------------------------------------------------------
  cvt_bf16_k<<<(NNODES * HDIM + 255) / 256, blk, 0, stream>>>(x, xbf,
                                                              NNODES * HDIM);
  transpose_w_k<<<(HDIM * HDIM + 255) / 256, blk, 0, stream>>>(Wq, Wqt, HDIM, HDIM);
  transpose_w_k<<<(HDIM * HDIM + 255) / 256, blk, 0, stream>>>(Wk, Wkt, HDIM, HDIM);
  transpose_w_k<<<(HDIM * HDIM + 255) / 256, blk, 0, stream>>>(Wv, Wvt, HDIM, HDIM);
  transpose_w_k<<<(HDIM * HDIM + 255) / 256, blk, 0, stream>>>(Wo, Wot, HDIM, HDIM);
  transpose_w_k<<<(2 * HDIM * HDIM + 255) / 256, blk, 0, stream>>>(Wm1, Wm1t, 2 * HDIM, HDIM);
  transpose_w_k<<<(HDIM * HDIM + 255) / 256, blk, 0, stream>>>(Wm2, Wm2t, HDIM, HDIM);
  transpose_w_k<<<(2 * HDIM * HDIM + 255) / 256, blk, 0, stream>>>(Wu1, Wu1t, 2 * HDIM, HDIM);
  transpose_w_k<<<(HDIM * HDIM + 255) / 256, blk, 0, stream>>>(Wu2, Wu2t, HDIM, HDIM);

  // --- QKV projections (scatter to per-head layouts) ------------------------
  const dim3 gProj(NNODES / 128, HDIM / 64);
  gemm_k<0, 4><<<gProj, blk, smem256, stream>>>(xbf, nullptr, nullptr, Wqt, bq,
                                                qbuf, NNODES, HDIM, QSCALE);
  gemm_k<0, 4><<<gProj, blk, smem256, stream>>>(xbf, nullptr, nullptr, Wkt, bk,
                                                kbuf, NNODES, HDIM, 1.0f);
  gemm_k<0, 5><<<gProj, blk, smem256, stream>>>(xbf, nullptr, nullptr, Wvt, bv,
                                                vTbuf, NNODES, HDIM, 1.0f);

  // --- flash attention ------------------------------------------------------
  attn_k<<<dim3(NNODES / 128, NHEADS), blk, 0, stream>>>(qbuf, kbuf, vTbuf,
                                                         attpre);

  // --- output projection ----------------------------------------------------
  gemm_k<0, 0><<<gProj, blk, smem256, stream>>>(attpre, nullptr, nullptr, Wot,
                                                bo, att, NNODES, HDIM, 1.0f);

  // --- edge MLP layer 1 (fused gather att[src]||att[dst]) -------------------
  const dim3 gEdge(NEDGES / 128, HDIM / 64);
  gemm_k<1, 1><<<gEdge, blk, smem512, stream>>>(att, nullptr, ei, Wm1t, bm1,
                                                hid, NEDGES, 2 * HDIM, 1.0f);

  // --- edge MLP layer 2 + segment-sum via atomics ---------------------------
  zero_f32_k<<<(NNODES * HDIM + 255) / 256, blk, 0, stream>>>(msg,
                                                              NNODES * HDIM);
  gemm_k<0, 3><<<gEdge, blk, smem256, stream>>>(hid, nullptr, ei, Wm2t, bm2,
                                                msg, NEDGES, HDIM, 1.0f);
  cvt_bf16_k<<<(NNODES * HDIM + 255) / 256, blk, 0, stream>>>(msg, msgbf,
                                                              NNODES * HDIM);

  // --- update MLP (fused concat att||messages), layer 2 -> d_out (f32) ------
  gemm_k<2, 1><<<gProj, blk, smem512, stream>>>(att, msgbf, nullptr, Wu1t, bu1,
                                                hid, NNODES, 2 * HDIM, 1.0f);
  gemm_k<0, 2><<<gProj, blk, smem256, stream>>>(hid, nullptr, nullptr, Wu2t,
                                                bu2, d_out, NNODES, HDIM, 1.0f);
}
